// SentenceGraphModel_1726576853059
// MI455X (gfx1250) — matrix operand order repaired and checked by reading, not compiled, not run on hardware
//
#include <hip/hip_runtime.h>
#include <cstdint>
#include <cstddef>

// Problem constants (match reference)
#define NROWS 16384
#define NEDGE 524288
#define IND   384
#define HD    256
#define HGD   128
#define OUTD  128
#define NLAY  3
#define NGR   64
#define EPSF  1e-16f

typedef __bf16 bf16;
typedef __attribute__((ext_vector_type(16))) __bf16 v16bf;
typedef __attribute__((ext_vector_type(8)))  float  v8f;

// CDNA5 async global->LDS path (guarded so either toolchain compiles)
#if __has_builtin(__builtin_amdgcn_global_load_async_to_lds_b128) && \
    __has_builtin(__builtin_amdgcn_s_wait_asynccnt)
#define USE_ASYNC_LDS 1
#else
#define USE_ASYNC_LDS 0
#endif

// Builtin expects: (int4 addrspace(1)*, int4 addrspace(3)*, imm int, imm int)
typedef int vint4 __attribute__((ext_vector_type(4)));
typedef __attribute__((address_space(1))) vint4 gvint4;
typedef __attribute__((address_space(3))) vint4 svint4;

// ---------------- helpers ----------------
__device__ __forceinline__ unsigned fenc(float f) {
    unsigned u = __float_as_uint(f);
    return (u & 0x80000000u) ? ~u : (u | 0x80000000u);
}
__device__ __forceinline__ float fdec(unsigned u) {
    u = (u & 0x80000000u) ? (u & 0x7fffffffu) : ~u;
    return __uint_as_float(u);
}

// ---------------- elementwise kernels ----------------
__global__ void k_cast_bf16(const float* __restrict__ in, bf16* __restrict__ out, int n) {
    int i = blockIdx.x * blockDim.x + threadIdx.x;
    if (i < n) out[i] = (bf16)in[i];
}
__global__ void k_fill_f32(float* p, float v, int n) {
    int i = blockIdx.x * blockDim.x + threadIdx.x;
    if (i < n) p[i] = v;
}
__global__ void k_fill_u32(unsigned* p, unsigned v, int n) {
    int i = blockIdx.x * blockDim.x + threadIdx.x;
    if (i < n) p[i] = v;
}
__global__ void k_swiglu(const float* __restrict__ a, const float* __restrict__ g,
                         float* __restrict__ h, int n) {
    int i = blockIdx.x * blockDim.x + threadIdx.x;
    if (i < n) {
        float x = g[i];
        h[i] = a[i] * (x / (1.0f + __expf(-x)));
    }
}
__global__ void k_rowdiv(const float* __restrict__ num, const float* __restrict__ den,
                         float* __restrict__ out, int cols, int n) {
    int i = blockIdx.x * blockDim.x + threadIdx.x;
    if (i < n) out[i] = num[i] / (den[i / cols] + EPSF);
}
__global__ void k_decode_max(const unsigned* __restrict__ menc, float* __restrict__ m, int n) {
    int i = blockIdx.x * blockDim.x + threadIdx.x;
    if (i < n) {
        float v = fdec(menc[i]);
        m[i] = (fabsf(v) < 3.0e38f) ? v : 0.0f;
    }
}

// ---------------- WMMA GEMM: C[M,Nn] = A[M,K] * B (+bias) ----------------
// BLAYOUT 0: B is [K,Nn] row-major (x @ W).  BLAYOUT 1: B is [Nn,K] row-major (x @ W^T).
// EPI 0: C = result (+bias if given).  EPI 2: C = relu(C + result (+bias)).
// Double-buffered LDS tiles; async global->LDS copies when available.
#define TM 128
#define TN 64
#define TK 32

template <int BLAYOUT, int EPI>
__global__ __launch_bounds__(256)
void k_wmma_gemm(const bf16* __restrict__ A, const bf16* __restrict__ B,
                 const float* __restrict__ bias, float* __restrict__ C,
                 int M, int Nn, int K) {
    __shared__ bf16 lds_a[2][TM * TK];   // [m][k]
    __shared__ bf16 lds_b[2][TN * TK];   // [n][k]

    const int bm   = blockIdx.x * TM;
    const int bn   = blockIdx.y * TN;
    const int tid  = threadIdx.x;
    const int wave = tid >> 5;
    const int lane = tid & 31;
    const int wm   = (wave & 3) * 32;   // wave tile row offset
    const int wn   = (wave >> 2) * 32;  // wave tile col offset
    const int l15  = lane & 15;
    const int ka   = (lane < 16) ? 0 : 16;   // A fragment K base
    const int kb   = (lane < 16) ? 0 : 16;   // B fragment K base

    v8f acc[2][2] = {};

    // Issue global->LDS staging of one K-tile into buffer `buf`.
    auto stage = [&](int k0, int buf) {
        // A tile: 128x32 bf16, 16B per chunk
        for (int c = tid; c < (TM * TK) / 8; c += 256) {
            int row = c >> 2;
            int col = (c & 3) * 8;
            bf16* ldst = &lds_a[buf][row * TK + col];
            if (bm + row < M) {
                const bf16* g = A + (size_t)(bm + row) * K + k0 + col;
#if USE_ASYNC_LDS
                __builtin_amdgcn_global_load_async_to_lds_b128(
                    (gvint4*)(const void*)g, (svint4*)(void*)ldst, 0, 0);
#else
                *(uint4*)ldst = *(const uint4*)g;
#endif
            } else {
                *(uint4*)ldst = make_uint4(0u, 0u, 0u, 0u);
            }
        }
        // B tile into [n][k]
        if (BLAYOUT == 1) {
            for (int c = tid; c < (TN * TK) / 8; c += 256) {
                int n  = c >> 2;
                int k8 = (c & 3) * 8;
                const bf16* g = B + (size_t)(bn + n) * K + k0 + k8;
                bf16* ldst = &lds_b[buf][n * TK + k8];
#if USE_ASYNC_LDS
                __builtin_amdgcn_global_load_async_to_lds_b128(
                    (gvint4*)(const void*)g, (svint4*)(void*)ldst, 0, 0);
#else
                *(uint4*)ldst = *(const uint4*)g;
#endif
            }
        } else {
            for (int c = tid; c < (TN * TK) / 8; c += 256) {
                int k  = c >> 3;
                int n8 = (c & 7) * 8;
                uint4 v = *(const uint4*)(B + (size_t)(k0 + k) * Nn + bn + n8);
                const bf16* pv = (const bf16*)&v;
#pragma unroll
                for (int j = 0; j < 8; ++j) lds_b[buf][(n8 + j) * TK + k] = pv[j];
            }
#if !USE_ASYNC_LDS
            if (k0 + TK < K && bm + (tid >> 1) < M)
                __builtin_prefetch(A + (size_t)(bm + (tid >> 1)) * K + k0 + TK, 0, 0);
#endif
        }
    };

    stage(0, 0);
    int buf = 0;
    for (int k0 = 0; k0 < K; k0 += TK) {
#if USE_ASYNC_LDS
        __builtin_amdgcn_s_wait_asynccnt(0);   // our piece of tile `buf` landed in LDS
#endif
        __syncthreads();                        // whole tile `buf` visible to all waves
        if (k0 + TK < K) stage(k0 + TK, buf ^ 1);   // overlap next tile with compute

        // ---- B fragments for this wave ----
        union { uint32_t u[8]; v16bf v; } bfr[2];
#pragma unroll
        for (int fj = 0; fj < 2; ++fj) {
            int bcol = wn + fj * 16 + l15;
            const uint32_t* bp = (const uint32_t*)&lds_b[buf][bcol * TK];
#pragma unroll
            for (int j = 0; j < 8; ++j) bfr[fj].u[j] = bp[kb / 2 + j];
        }
        // ---- A fragments + WMMA ----
#pragma unroll
        for (int fi = 0; fi < 2; ++fi) {
            int arow = wm + fi * 16 + l15;
            union { uint32_t u[8]; v16bf v; } afr;
            const uint32_t* ap = (const uint32_t*)&lds_a[buf][arow * TK];
#pragma unroll
            for (int j = 0; j < 4; ++j) {
                afr.u[j]     = ap[ka / 2 + j];
                afr.u[4 + j] = ap[ka / 2 + 4 + j];
            }
#pragma unroll
            for (int fj = 0; fj < 2; ++fj) {
                acc[fi][fj] = __builtin_amdgcn_wmma_f32_16x16x32_bf16(
                    false, afr.v, false, bfr[fj].v, (short)0, acc[fi][fj], false, false);
            }
        }
        buf ^= 1;
    }

    // ---- epilogue: f32 16x16 C layout: lane holds col (lane&15), vgpr v -> row v(+8) ----
    const int rbase = (lane < 16) ? 0 : 8;
#pragma unroll
    for (int fi = 0; fi < 2; ++fi) {
#pragma unroll
        for (int fj = 0; fj < 2; ++fj) {
#pragma unroll
            for (int v = 0; v < 8; ++v) {
                int m = bm + wm + fi * 16 + rbase + v;
                int n = bn + wn + fj * 16 + l15;
                if (m < M) {
                    float val = acc[fi][fj][v];
                    if (bias) val += bias[n];
                    size_t off = (size_t)m * Nn + n;
                    if (EPI == 2) {
                        val += C[off];
                        val = fmaxf(val, 0.0f);
                    }
                    C[off] = val;
                }
            }
        }
    }
}

// ---------------- GRU sequential scan (both directions concurrently) ----------------
__global__ __launch_bounds__(384)
void k_gru_scan(const float* __restrict__ giF, const float* __restrict__ giB,
                const float* __restrict__ WhhF, const float* __restrict__ WhhB,
                const float* __restrict__ bhhF, const float* __restrict__ bhhB,
                float* __restrict__ out) {
    const int dir = blockIdx.x;   // 0 = forward, 1 = backward
    const float* gi  = dir ? giB  : giF;
    const float* Whh = dir ? WhhB : WhhF;
    const float* bhh = dir ? bhhB : bhhF;
    const int col_off = dir ? HGD : 0;

    __shared__ float hprev[HGD];
    __shared__ float gh[3 * HGD];
    const int tid = threadIdx.x;
    if (tid < HGD) hprev[tid] = 0.0f;
    __syncthreads();

    const float* wr = Whh + (size_t)tid * HGD;
    const float  bh = bhh[tid];

    for (int s = 0; s < NROWS; ++s) {
        int t = dir ? (NROWS - 1 - s) : s;
        float acc = 0.0f;
#pragma unroll 4
        for (int k = 0; k < HGD; ++k) acc += wr[k] * hprev[k];
        gh[tid] = acc + bh;
        __syncthreads();
        if (tid < HGD) {
            const float* g = gi + (size_t)t * (3 * HGD);
            float r  = 1.0f / (1.0f + __expf(-(g[tid] + gh[tid])));
            float z  = 1.0f / (1.0f + __expf(-(g[HGD + tid] + gh[HGD + tid])));
            float nn = tanhf(g[2 * HGD + tid] + r * gh[2 * HGD + tid]);
            float hn = (1.0f - z) * nn + z * hprev[tid];
            out[(size_t)t * HD + col_off + tid] = hn;
            hprev[tid] = hn;
        }
        __syncthreads();
    }
}

// ---------------- GAT / prop kernels ----------------
__global__ void k_gat_scores(const float* __restrict__ hW, const float* __restrict__ a_src,
                             const float* __restrict__ a_dst,
                             float* __restrict__ ssrc, float* __restrict__ sdst) {
    int row  = blockIdx.x * (blockDim.x >> 5) + (threadIdx.x >> 5);
    int lane = threadIdx.x & 31;
    if (row >= NROWS) return;
    const float* hr = hW + (size_t)row * HD;
    float s1 = 0.0f, s2 = 0.0f;
    for (int c = lane; c < HD; c += 32) {
        float v = hr[c];
        s1 += v * a_src[c];
        s2 += v * a_dst[c];
    }
    for (int off = 16; off; off >>= 1) {
        s1 += __shfl_down(s1, off, 32);
        s2 += __shfl_down(s2, off, 32);
    }
    if (lane == 0) { ssrc[row] = s1; sdst[row] = s2; }
}

__global__ void k_gat_edge_max(const int* __restrict__ src, const int* __restrict__ dst,
                               const float* __restrict__ ssrc, const float* __restrict__ sdst,
                               float* __restrict__ ebuf, unsigned* __restrict__ menc) {
    int e = blockIdx.x * blockDim.x + threadIdx.x;
    if (e >= NEDGE) return;
    int s = src[e], d = dst[e];
    float v = ssrc[s] + sdst[d];
    v = (v > 0.0f) ? v : 0.2f * v;   // leaky_relu 0.2
    ebuf[e] = v;
    atomicMax(&menc[d], fenc(v));
}

__global__ void k_gat_edge_scatter(const int* __restrict__ src, const int* __restrict__ dst,
                                   const float* __restrict__ ebuf, const float* __restrict__ m,
                                   const float* __restrict__ ew, const float* __restrict__ hW,
                                   float* __restrict__ denom, float* __restrict__ msg) {
    int e    = blockIdx.x * (blockDim.x >> 5) + (threadIdx.x >> 5);
    int lane = threadIdx.x & 31;
    if (e >= NEDGE) return;
    int s = src[e], d = dst[e];
    float w = __expf(ebuf[e] - m[d]) * ew[e];
    if (lane == 0) atomicAdd(&denom[d], w);
    const float* hs = hW + (size_t)s * HD;
    float* md = msg + (size_t)d * HD;
    for (int c = lane; c < HD; c += 32) atomicAdd(&md[c], w * hs[c]);
}

__global__ void k_prop_edge_scatter(const int* __restrict__ src, const int* __restrict__ dst,
                                    const float* __restrict__ ew, const float* __restrict__ x,
                                    float* __restrict__ num, float* __restrict__ den) {
    int e    = blockIdx.x * (blockDim.x >> 5) + (threadIdx.x >> 5);
    int lane = threadIdx.x & 31;
    if (e >= NEDGE) return;
    int s = src[e], d = dst[e];
    float w = ew[e];
    if (lane == 0) atomicAdd(&den[d], w);
    const float* xs = x + (size_t)s * HD;
    float* nd = num + (size_t)d * HD;
    for (int c = lane; c < HD; c += 32) atomicAdd(&nd[c], w * xs[c]);
}

// ---------------- readout ----------------
__global__ void k_readout_acc(const float* __restrict__ h, const long long* __restrict__ batch,
                              float* __restrict__ cnt, float* __restrict__ sum,
                              unsigned* __restrict__ mxe) {
    int row  = blockIdx.x * (blockDim.x >> 5) + (threadIdx.x >> 5);
    int lane = threadIdx.x & 31;
    if (row >= NROWS) return;
    int g = (int)batch[row];
    if (lane == 0) atomicAdd(&cnt[g], 1.0f);
    const float* hr = h + (size_t)row * HD;
    for (int c = lane; c < HD; c += 32) {
        float v = hr[c];
        atomicAdd(&sum[(size_t)g * HD + c], v);
        atomicMax(&mxe[(size_t)g * HD + c], fenc(v));
    }
}

__global__ void k_readout_build(const float* __restrict__ sum, const unsigned* __restrict__ mxe,
                                const float* __restrict__ cnt, float* __restrict__ r) {
    int i = blockIdx.x * blockDim.x + threadIdx.x;
    if (i >= NGR * HD) return;
    int g = i / HD, c = i % HD;
    r[(size_t)g * (2 * HD) + c] = sum[i] / (cnt[g] + EPSF);
    float mv = fdec(mxe[i]);
    r[(size_t)g * (2 * HD) + HD + c] = (fabsf(mv) < 3.0e38f) ? mv : 0.0f;
}

// ---------------- host-side orchestration ----------------
static inline char* align256(char* p) {
    return (char*)(((uintptr_t)p + 255u) & ~(uintptr_t)255u);
}

static void cast_bf16(const float* in, bf16* out, int n, hipStream_t s) {
    k_cast_bf16<<<(n + 255) / 256, 256, 0, s>>>(in, out, n);
}
static void run_gemm(int blay, int epi, const bf16* A, const bf16* B, const float* bias,
                     float* C, int M, int Nn, int K, hipStream_t s) {
    dim3 g((M + TM - 1) / TM, Nn / TN), b(256);
    if (blay == 0 && epi == 0)      k_wmma_gemm<0, 0><<<g, b, 0, s>>>(A, B, bias, C, M, Nn, K);
    else if (blay == 1 && epi == 0) k_wmma_gemm<1, 0><<<g, b, 0, s>>>(A, B, bias, C, M, Nn, K);
    else                            k_wmma_gemm<0, 2><<<g, b, 0, s>>>(A, B, bias, C, M, Nn, K);
}

extern "C" void kernel_launch(void* const* d_in, const int* in_sizes, int n_in,
                              void* d_out, int out_size, void* d_ws, size_t ws_size,
                              hipStream_t stream) {
    (void)in_sizes; (void)n_in; (void)out_size; (void)ws_size;
    // ---- inputs (setup_inputs dict order; params flattened in insertion order) ----
    int idx = 0;
    const float*     x  = (const float*)d_in[idx++];
    const int*       ei = (const int*)d_in[idx++];
    const float*     ew = (const float*)d_in[idx++];
    const long long* bt = (const long long*)d_in[idx++];
    const int* src = ei;
    const int* dst = ei + NEDGE;

    const float* in_w  = (const float*)d_in[idx++];
    const float* in_b  = (const float*)d_in[idx++];
    const float* sw_w  = (const float*)d_in[idx++];
    const float* sw_b  = (const float*)d_in[idx++];
    const float* sv_w  = (const float*)d_in[idx++];
    const float* sv_b  = (const float*)d_in[idx++];
    const float* out_w = (const float*)d_in[idx++];
    const float* out_b = (const float*)d_in[idx++];
    const float *Wih[NLAY][2], *Whh[NLAY][2], *bih[NLAY][2], *bhh[NLAY][2];
    for (int l = 0; l < NLAY; ++l)
        for (int d = 0; d < 2; ++d) {
            Wih[l][d] = (const float*)d_in[idx++];
            Whh[l][d] = (const float*)d_in[idx++];
            bih[l][d] = (const float*)d_in[idx++];
            bhh[l][d] = (const float*)d_in[idx++];
        }
    const float *gat_w[NLAY], *gat_b[NLAY], *gat_as[NLAY], *gat_ad[NLAY];
    for (int l = 0; l < NLAY; ++l) {
        gat_w[l]  = (const float*)d_in[idx++];
        gat_b[l]  = (const float*)d_in[idx++];
        gat_as[l] = (const float*)d_in[idx++];
        gat_ad[l] = (const float*)d_in[idx++];
    }
    const float *pw_self[NLAY], *pw_nbr[NLAY], *p_b[NLAY];
    for (int l = 0; l < NLAY; ++l) {
        pw_self[l] = (const float*)d_in[idx++];
        pw_nbr[l]  = (const float*)d_in[idx++];
        p_b[l]     = (const float*)d_in[idx++];
    }

    // ---- workspace carve-up ----
    char* p = align256((char*)d_ws);
    float* h   = (float*)p;               p = align256(p + (size_t)NROWS * HD * 4);
    float* t1  = (float*)p;               p = align256(p + (size_t)NROWS * IND * 4);
    float* t2  = (float*)p;               p = align256(p + (size_t)NROWS * IND * 4);
    float* t3  = (float*)p;               p = align256(p + (size_t)NROWS * HD * 4);
    bf16*  hb  = (bf16*)p;                p = align256(p + (size_t)NROWS * IND * 2);
    bf16*  hb2 = (bf16*)p;                p = align256(p + (size_t)NROWS * HD * 2);
    float* ebuf = (float*)p;              p = align256(p + (size_t)NEDGE * 4);
    float* ssrc = (float*)p;              p = align256(p + (size_t)NROWS * 4);
    float* sdst = (float*)p;              p = align256(p + (size_t)NROWS * 4);
    unsigned* menc = (unsigned*)p;        p = align256(p + (size_t)NROWS * 4);
    float* denom = (float*)p;             p = align256(p + (size_t)NROWS * 4);
    float* cnt  = (float*)p;              p = align256(p + (size_t)NGR * 4);
    float* gsum = (float*)p;              p = align256(p + (size_t)NGR * HD * 4);
    unsigned* gmx = (unsigned*)p;         p = align256(p + (size_t)NGR * HD * 4);
    float* rbuf = (float*)p;              p = align256(p + (size_t)NGR * 2 * HD * 4);
    bf16*  rb   = (bf16*)p;               p = align256(p + (size_t)NGR * 2 * HD * 2);
    // bf16 weight arena
    bf16* in_wb  = (bf16*)p;              p = align256(p + (size_t)IND * HD * 2);
    bf16* sw_wb  = (bf16*)p;              p = align256(p + (size_t)HD * HD * 2);
    bf16* sv_wb  = (bf16*)p;              p = align256(p + (size_t)HD * HD * 2);
    bf16* out_wb = (bf16*)p;              p = align256(p + (size_t)2 * HD * OUTD * 2);
    bf16* Wihb[NLAY][2];
    for (int l = 0; l < NLAY; ++l)
        for (int d = 0; d < 2; ++d) { Wihb[l][d] = (bf16*)p; p = align256(p + (size_t)3 * HGD * HD * 2); }
    bf16* gat_wb[NLAY];
    for (int l = 0; l < NLAY; ++l) { gat_wb[l] = (bf16*)p; p = align256(p + (size_t)HD * HD * 2); }
    bf16 *pw_selfb[NLAY], *pw_nbrb[NLAY];
    for (int l = 0; l < NLAY; ++l) {
        pw_selfb[l] = (bf16*)p;           p = align256(p + (size_t)HD * HD * 2);
        pw_nbrb[l]  = (bf16*)p;           p = align256(p + (size_t)HD * HD * 2);
    }

    const unsigned NEG_INF_ENC = 0x007fffffu;   // fenc(-inf)

    // ---- 0) convert weights to bf16 ----
    cast_bf16(in_w, in_wb, IND * HD, stream);
    cast_bf16(sw_w, sw_wb, HD * HD, stream);
    cast_bf16(sv_w, sv_wb, HD * HD, stream);
    cast_bf16(out_w, out_wb, 2 * HD * OUTD, stream);
    for (int l = 0; l < NLAY; ++l) {
        for (int d = 0; d < 2; ++d) cast_bf16(Wih[l][d], Wihb[l][d], 3 * HGD * HD, stream);
        cast_bf16(gat_w[l], gat_wb[l], HD * HD, stream);
        cast_bf16(pw_self[l], pw_selfb[l], HD * HD, stream);
        cast_bf16(pw_nbr[l], pw_nbrb[l], HD * HD, stream);
    }

    // ---- 1) input projection: h = x @ in_w + in_b ----
    cast_bf16(x, hb, NROWS * IND, stream);
    run_gemm(0, 0, hb, in_wb, in_b, h, NROWS, HD, IND, stream);

    // ---- 2) BiGRU + SwiGLU stack ----
    for (int l = 0; l < NLAY; ++l) {
        cast_bf16(h, hb, NROWS * HD, stream);
        // gi = h @ Wih^T + bih  (Wih stored [3HG, H] -> NK layout)
        run_gemm(1, 0, hb, Wihb[l][0], bih[l][0], t1, NROWS, 3 * HGD, HD, stream);
        run_gemm(1, 0, hb, Wihb[l][1], bih[l][1], t2, NROWS, 3 * HGD, HD, stream);
        // sequential recurrences, both directions at once; writes h in place
        k_gru_scan<<<2, 384, 0, stream>>>(t1, t2, Whh[l][0], Whh[l][1],
                                          bhh[l][0], bhh[l][1], h);
        // SwiGLU
        cast_bf16(h, hb, NROWS * HD, stream);
        run_gemm(0, 0, hb, sw_wb, sw_b, t1, NROWS, HD, HD, stream);
        run_gemm(0, 0, hb, sv_wb, sv_b, t3, NROWS, HD, HD, stream);
        k_swiglu<<<(NROWS * HD + 255) / 256, 256, 0, stream>>>(t1, t3, h, NROWS * HD);
    }

    // ---- 3) GAT + propagation stack ----
    for (int l = 0; l < NLAY; ++l) {
        // hW = h @ w + b  -> t1
        cast_bf16(h, hb, NROWS * HD, stream);
        run_gemm(0, 0, hb, gat_wb[l], gat_b[l], t1, NROWS, HD, HD, stream);
        // per-node attention scores
        k_gat_scores<<<NROWS / 8, 256, 0, stream>>>(t1, gat_as[l], gat_ad[l], ssrc, sdst);
        // segment max over edges
        k_fill_u32<<<(NROWS + 255) / 256, 256, 0, stream>>>(menc, NEG_INF_ENC, NROWS);
        k_gat_edge_max<<<NEDGE / 256, 256, 0, stream>>>(src, dst, ssrc, sdst, ebuf, menc);
        k_decode_max<<<(NROWS + 255) / 256, 256, 0, stream>>>(menc, sdst, NROWS); // sdst := m
        // softmax-weighted message scatter
        k_fill_f32<<<(NROWS + 255) / 256, 256, 0, stream>>>(denom, 0.0f, NROWS);
        k_fill_f32<<<(NROWS * HD + 255) / 256, 256, 0, stream>>>(t2, 0.0f, NROWS * HD);
        k_gat_edge_scatter<<<NEDGE / 8, 256, 0, stream>>>(src, dst, ebuf, sdst, ew, t1,
                                                          denom, t2);
        k_rowdiv<<<(NROWS * HD + 255) / 256, 256, 0, stream>>>(t2, denom, t3, HD, NROWS * HD);
        // prop: weighted-mean neighbor aggregation of a (=t3)
        k_fill_f32<<<(NROWS + 255) / 256, 256, 0, stream>>>(denom, 0.0f, NROWS);
        k_fill_f32<<<(NROWS * HD + 255) / 256, 256, 0, stream>>>(t2, 0.0f, NROWS * HD);
        k_prop_edge_scatter<<<NEDGE / 8, 256, 0, stream>>>(src, dst, ew, t3, t2, denom);
        k_rowdiv<<<(NROWS * HD + 255) / 256, 256, 0, stream>>>(t2, denom, t1, HD, NROWS * HD);
        // h = relu(a @ w_self + agg @ w_nbr + b)
        cast_bf16(t3, hb, NROWS * HD, stream);   // a
        cast_bf16(t1, hb2, NROWS * HD, stream);  // agg
        run_gemm(0, 0, hb, pw_selfb[l], p_b[l], h, NROWS, HD, HD, stream);
        run_gemm(0, 2, hb2, pw_nbrb[l], nullptr, h, NROWS, HD, HD, stream); // add+relu
    }

    // ---- 4) readout: per-graph mean+max -> [NG, 2H] -> out proj ----
    k_fill_f32<<<1, NGR, 0, stream>>>(cnt, 0.0f, NGR);
    k_fill_f32<<<(NGR * HD + 255) / 256, 256, 0, stream>>>(gsum, 0.0f, NGR * HD);
    k_fill_u32<<<(NGR * HD + 255) / 256, 256, 0, stream>>>(gmx, NEG_INF_ENC, NGR * HD);
    k_readout_acc<<<NROWS / 8, 256, 0, stream>>>(h, bt, cnt, gsum, gmx);
    k_readout_build<<<(NGR * HD + 255) / 256, 256, 0, stream>>>(gsum, gmx, cnt, rbuf);
    cast_bf16(rbuf, rb, NGR * 2 * HD, stream);
    run_gemm(0, 0, rb, out_wb, out_b, (float*)d_out, NGR, OUTD, 2 * HD, stream);
}
